// PointNetSetAbstractionMSG_20675972563784
// MI455X (gfx1250) — matrix operand order repaired and checked by reading, not compiled, not run on hardware
//
#include <hip/hip_runtime.h>
#include <hip/hip_bf16.h>

typedef __attribute__((ext_vector_type(16))) _Float16 v16h;
typedef __attribute__((ext_vector_type(8)))  _Float16 v8h;
typedef __attribute__((ext_vector_type(8)))  float    v8f;

#define BB   16
#define NN   4096
#define SS   1024
#define EPSF 1e-5f
#define OUTC 320
#define KPAD1 96   // layer-1 input channels (67) padded to multiple of 32
#define LDSPAD 8   // bank-conflict padding (halves) for the LDS weight panel

// ---------------------------------------------------------------------------
// 1) Farthest point sampling: one block per batch, 256 threads, each thread
//    owns 16 points (cached in registers). Sequential 1024-step scan with
//    LDS argmax reduction; tie-break = smallest index (matches jnp.argmax).
//    Emits current `far` BEFORE updating (matches lax.scan output).
// ---------------------------------------------------------------------------
__global__ void fps_kernel(const float* __restrict__ xyz,      // (B,3,N)
                           int* __restrict__ fps_idx,          // (B,S)
                           float* __restrict__ nxyz,           // (B,S,3)
                           float* __restrict__ out_xyz)        // (B,3,S) -> d_out
{
  const int b   = blockIdx.x;
  const int tid = threadIdx.x;
  const float* xb = xyz + (size_t)b * 3 * NN;

  float dist[NN / 256];
  float px[NN / 256], py[NN / 256], pz[NN / 256];
#pragma unroll
  for (int i = 0; i < NN / 256; i++) {
    int n = i * 256 + tid;
    dist[i] = 1e10f;
    px[i] = xb[n];
    py[i] = xb[NN + n];
    pz[i] = xb[2 * NN + n];
  }

  __shared__ float sval[256];
  __shared__ int   sidx[256];
  __shared__ float cx, cy, cz;
  __shared__ int   sfar;
  if (tid == 0) sfar = 0;
  __syncthreads();

  for (int s = 0; s < SS; s++) {
    if (tid == 0) {
      int far = sfar;
      float fx = xb[far], fy = xb[NN + far], fz = xb[2 * NN + far];
      cx = fx; cy = fy; cz = fz;
      fps_idx[b * SS + s] = far;
      size_t nb = ((size_t)b * SS + s) * 3;
      nxyz[nb + 0] = fx; nxyz[nb + 1] = fy; nxyz[nb + 2] = fz;
      out_xyz[((size_t)b * 3 + 0) * SS + s] = fx;
      out_xyz[((size_t)b * 3 + 1) * SS + s] = fy;
      out_xyz[((size_t)b * 3 + 2) * SS + s] = fz;
    }
    __syncthreads();

    float qx = cx, qy = cy, qz = cz;
    float bv = -1.0f; int bi = 0;
#pragma unroll
    for (int i = 0; i < NN / 256; i++) {
      float dx = px[i] - qx, dy = py[i] - qy, dz = pz[i] - qz;
      float d  = dx * dx + dy * dy + dz * dz;
      float nd = fminf(dist[i], d);
      dist[i]  = nd;
      if (nd > bv) { bv = nd; bi = i * 256 + tid; }
    }
    sval[tid] = bv; sidx[tid] = bi;
    __syncthreads();
    for (int off = 128; off >= 1; off >>= 1) {
      if (tid < off) {
        float ov = sval[tid + off]; int oi = sidx[tid + off];
        if (ov > sval[tid] || (ov == sval[tid] && oi < sidx[tid])) {
          sval[tid] = ov; sidx[tid] = oi;
        }
      }
      __syncthreads();
    }
    if (tid == 0) sfar = sidx[0];
    __syncthreads();
  }
}

// ---------------------------------------------------------------------------
// 2) Ball query: one wave32 per (b,s) query; ballot compaction keeps the
//    first K indices (ascending) with d^2 <= r^2, pads with the first hit.
// ---------------------------------------------------------------------------
__global__ void ballquery_kernel(const float* __restrict__ xyz,   // (B,3,N)
                                 const float* __restrict__ nxyz,  // (B,S,3)
                                 int* __restrict__ gidx,          // (B*S,K)
                                 float r2, int K)
{
  int gw   = (int)((blockIdx.x * blockDim.x + threadIdx.x) >> 5);
  int lane = threadIdx.x & 31;
  if (gw >= BB * SS) return;
  int b = gw / SS;
  const float* xb = xyz + (size_t)b * 3 * NN;
  float qx = nxyz[(size_t)gw * 3 + 0];
  float qy = nxyz[(size_t)gw * 3 + 1];
  float qz = nxyz[(size_t)gw * 3 + 2];
  int* out = gidx + (size_t)gw * K;

  int cnt = 0, first = -1;
  for (int j0 = 0; j0 < NN && cnt < K; j0 += 32) {
    int j = j0 + lane;
    float dx = xb[j] - qx, dy = xb[NN + j] - qy, dz = xb[2 * NN + j] - qz;
    float d2 = dx * dx + dy * dy + dz * dz;
    unsigned mask = __builtin_amdgcn_ballot_w32(d2 <= r2);
    if (first < 0 && mask) first = j0 + __builtin_ctz(mask);
    if (d2 <= r2) {
      int pos = cnt + __builtin_popcount(mask & ((1u << lane) - 1u));
      if (pos < K) out[pos] = j;
    }
    cnt += __builtin_popcount(mask);
  }
  if (cnt > K) cnt = K;
  if (first < 0) first = NN - 1;  // JAX gather clamps OOB
  for (int t = cnt + lane; t < K; t += 32) out[t] = first;
}

// ---------------------------------------------------------------------------
// 3) Gather grouped features directly as f16, row stride KPAD1, zero-padded:
//    Xh[r, 0..63] = points[b,c,idx]; Xh[r,64..66] = xyz - center; rest 0.
// ---------------------------------------------------------------------------
__global__ void gather_kernel(const float* __restrict__ points, // (B,64,N)
                              const float* __restrict__ xyz,    // (B,3,N)
                              const float* __restrict__ nxyz,   // (B,S,3)
                              const int* __restrict__ gidx,     // (B*S,K)
                              int K, _Float16* __restrict__ Xh) // (R,KPAD1)
{
  long total = (long)BB * SS * K * KPAD1;
  for (long idx = (long)blockIdx.x * blockDim.x + threadIdx.x; idx < total;
       idx += (long)gridDim.x * blockDim.x) {
    long r = idx / KPAD1;
    int  c = (int)(idx - r * KPAD1);
    long bs = r / K;
    int  b  = (int)(bs / SS);
    int  pi = gidx[r];
    float v;
    if (c < 64)      v = points[((size_t)b * 64 + c) * NN + pi];
    else if (c < 67) { int d = c - 64; v = xyz[((size_t)b * 3 + d) * NN + pi] - nxyz[bs * 3 + d]; }
    else             v = 0.0f;
    Xh[idx] = (_Float16)v;
  }
}

// ---------------------------------------------------------------------------
// 4) Weight prep: f32 (cout,cin) -> f16 (cout,kpad), zero-padded K.
// ---------------------------------------------------------------------------
__global__ void prep_w(const float* __restrict__ W, int cin, int kpad, int cout,
                       _Float16* __restrict__ Wh)
{
  int total = cout * kpad;
  for (int i = blockIdx.x * blockDim.x + threadIdx.x; i < total;
       i += gridDim.x * blockDim.x) {
    int n = i / kpad, k = i - n * kpad;
    Wh[i] = (k < cin) ? (_Float16)W[(size_t)n * cin + k] : (_Float16)0.0f;
  }
}

// ---------------------------------------------------------------------------
// 5) GEMM: Y[r,o] = sum_k Xh[r,k]*Wh[o,k] + bias[o].
//    Block = 4 waves x 16 rows; each wave computes ALL cout columns for its
//    rows (NG = cout/32 column groups, 2*NG accumulators), so the large X
//    stream is read from HBM exactly once. The full weight panel is staged
//    in LDS once per block (<=26KB, bank-padded) and B fragments come from
//    ds_load_b128. Fragment layouts per CDNA5 ISA 7.12.2.
// ---------------------------------------------------------------------------
template <int NG>
__global__ __launch_bounds__(128) void gemm_wmma(
    const _Float16* __restrict__ Xh, int kpad,
    const _Float16* __restrict__ Wh,
    const float* __restrict__ bias,
    float* __restrict__ Y)
{
  constexpr int cout = NG * 32;
  const int lds_k = kpad + LDSPAD;
  __shared__ __align__(32) _Float16 sW[128 * (KPAD1 + LDSPAD)];

  const int tid = threadIdx.x;
  // Cooperative weight-panel load: cout*kpad halves in 16B chunks.
  {
    const int chunks_per_row = kpad / 8;          // kpad multiple of 32
    const int total_chunks   = cout * chunks_per_row;
    for (int i = tid; i < total_chunks; i += 128) {
      int n  = i / chunks_per_row;
      int kk = (i - n * chunks_per_row) * 8;
      *(v8h*)(sW + n * lds_k + kk) = *(const v8h*)(Wh + (size_t)n * kpad + kk);
    }
  }
  __syncthreads();

  const int lane = tid & 31;
  const int wave = tid >> 5;
  const long r0  = ((long)blockIdx.x * 4 + wave) * 16;
  const int  half = lane >> 4;
  const int  lm   = lane & 15;

  v8f acc[2 * NG];
#pragma unroll
  for (int i = 0; i < 2 * NG; i++) acc[i] = (v8f){};

  const _Float16* xr = Xh + (size_t)(r0 + lm) * kpad;   // A row (m = lm)

  for (int kb = 0; kb < kpad; kb += 32) {
    // A (16x32 f16): e0..7 -> k = kb+half*8+e; e8..15 -> k = kb+16+half*8+(e-8)
    v8h alo = *(const v8h*)(xr + kb + half * 8);
    v8h ahi = *(const v8h*)(xr + kb + 16 + half * 8);
    v16h a  = __builtin_shufflevector(alo, ahi, 0, 1, 2, 3, 4, 5, 6, 7,
                                      8, 9, 10, 11, 12, 13, 14, 15);
#pragma unroll
    for (int g = 0; g < NG; g++) {
      // B (32x16 f16): lane holds column n = g*32+lm (and +16);
      // e0..7 -> k = kb+half*8+e; e8..15 -> k = kb+16+half*8+(e-8).
      const _Float16* b0p = sW + (g * 32 + lm) * lds_k + kb + half * 8;
      const _Float16* b1p = sW + (g * 32 + 16 + lm) * lds_k + kb + half * 8;
      v8h b0lo = *(const v8h*)(b0p);
      v8h b0hi = *(const v8h*)(b0p + 16);
      v8h b1lo = *(const v8h*)(b1p);
      v8h b1hi = *(const v8h*)(b1p + 16);
      v16h b0 = __builtin_shufflevector(b0lo, b0hi, 0, 1, 2, 3, 4, 5, 6, 7,
                                        8, 9, 10, 11, 12, 13, 14, 15);
      v16h b1 = __builtin_shufflevector(b1lo, b1hi, 0, 1, 2, 3, 4, 5, 6, 7,
                                        8, 9, 10, 11, 12, 13, 14, 15);
      acc[2 * g]     = __builtin_amdgcn_wmma_f32_16x16x32_f16(
          false, a, false, b0, (short)0, acc[2 * g], false, false);
      acc[2 * g + 1] = __builtin_amdgcn_wmma_f32_16x16x32_f16(
          false, a, false, b1, (short)0, acc[2 * g + 1], false, false);
    }
  }

#pragma unroll
  for (int g = 0; g < NG; g++) {
    const int n = g * 32 + lm;
    float bn0 = bias[n], bn1 = bias[n + 16];
#pragma unroll
    for (int v = 0; v < 8; v++) {
      long rr = (r0 + half * 8 + v) * cout;    // C/D layout: VGPR v -> row v (+8 hi)
      Y[rr + n]      = acc[2 * g][v] + bn0;
      Y[rr + n + 16] = acc[2 * g + 1][v] + bn1;
    }
  }
}

// ---------------------------------------------------------------------------
// 6) Deterministic per-channel stats (two-stage), folded into scale/shift.
// ---------------------------------------------------------------------------
__global__ void stats_partial(const float* __restrict__ Y, long R, int cout,
                              float* __restrict__ part)  // (nblk, 256)
{
  int c = threadIdx.x;              // blockDim = 128
  if (c >= cout) return;
  float s = 0.0f, sq = 0.0f;
  for (long r = blockIdx.x; r < R; r += gridDim.x) {
    float v = Y[r * cout + c];
    s += v; sq += v * v;
  }
  part[(size_t)blockIdx.x * 256 + c]       = s;
  part[(size_t)blockIdx.x * 256 + 128 + c] = sq;
}

__global__ void stats_finalize(const float* __restrict__ part, int nblk,
                               const float* __restrict__ gamma,
                               const float* __restrict__ beta,
                               long R, int cout,
                               float* __restrict__ scale, float* __restrict__ shift)
{
  int c = threadIdx.x;
  if (c >= cout) return;
  float s = 0.0f, sq = 0.0f;
  for (int k = 0; k < nblk; k++) {
    s  += part[(size_t)k * 256 + c];
    sq += part[(size_t)k * 256 + 128 + c];
  }
  float inv_n = 1.0f / (float)R;
  float mu  = s * inv_n;
  float var = sq * inv_n - mu * mu;
  float isd = rsqrtf(var + EPSF);
  float sc  = gamma[c] * isd;
  scale[c] = sc;
  shift[c] = beta[c] - mu * sc;
}

// ---------------------------------------------------------------------------
// 7) BN + ReLU applied once, emitted as f16 (next layer's A operand).
// ---------------------------------------------------------------------------
__global__ void norm_relu_f16(const float* __restrict__ Y, long total,
                              const float* __restrict__ scale,
                              const float* __restrict__ shift,
                              int cout, _Float16* __restrict__ Xh)
{
  for (long i = (long)blockIdx.x * blockDim.x + threadIdx.x; i < total;
       i += (long)gridDim.x * blockDim.x) {
    int c = (int)(i % cout);
    float v = fmaxf(Y[i] * scale[c] + shift[c], 0.0f);
    Xh[i] = (_Float16)v;
  }
}

// ---------------------------------------------------------------------------
// 8) Final BN + ReLU + max over K, scatter into (B, 320, S) output slice.
// ---------------------------------------------------------------------------
__global__ void maxpool_kernel(const float* __restrict__ Y,   // (R,cout)
                               const float* __restrict__ scale,
                               const float* __restrict__ shift,
                               int K, int cout,
                               float* __restrict__ outp, int chBase)
{
  long bs = blockIdx.x;            // 0 .. B*S-1
  int  c  = threadIdx.x;
  if (c >= cout) return;
  int b = (int)(bs / SS), s = (int)(bs % SS);
  const float* yb = Y + (size_t)bs * K * cout;
  float sc = scale[c], sh = shift[c];
  float m = 0.0f;
  for (int k = 0; k < K; k++) {
    float v = fmaxf(yb[(size_t)k * cout + c] * sc + sh, 0.0f);
    m = fmaxf(m, v);
  }
  outp[((size_t)b * OUTC + chBase + c) * SS + s] = m;
}

// ---------------------------------------------------------------------------
extern "C" void kernel_launch(void* const* d_in, const int* in_sizes, int n_in,
                              void* d_out, int out_size, void* d_ws, size_t ws_size,
                              hipStream_t stream)
{
  const float* xyz    = (const float*)d_in[0];
  const float* points = (const float*)d_in[1];
  // params flattened: d_in[2 + (branch*3 + layer)*4 + {0:W, 1:b, 2:gamma, 3:beta}]

  char* ws = (char*)d_ws;
  auto carve = [&](size_t bytes) -> char* {
    char* p = ws; ws += (bytes + 255) & ~(size_t)255; return p;
  };
  const long RMAX = (long)BB * SS * 128;
  int*      fps_idx = (int*)      carve((size_t)BB * SS * 4);
  float*    nxyz    = (float*)    carve((size_t)BB * SS * 3 * 4);
  int*      gidx    = (int*)      carve((size_t)BB * SS * 128 * 4);
  float*    part    = (float*)    carve((size_t)1024 * 256 * 4);
  float*    stats   = (float*)    carve((size_t)3 * 2 * 128 * 4);
  _Float16* Wh      = (_Float16*) carve((size_t)128 * KPAD1 * 2);
  _Float16* XhA     = (_Float16*) carve((size_t)RMAX * 96 * 2);
  _Float16* XhB     = (_Float16*) carve((size_t)RMAX * 64 * 2);
  float*    Yf      = (float*)    carve((size_t)RMAX * 128 * 4);

  float* out_xyz = (float*)d_out;                     // (B,3,S)
  float* out_pts = out_xyz + (size_t)BB * 3 * SS;     // (B,320,S)

  fps_kernel<<<BB, 256, 0, stream>>>(xyz, fps_idx, nxyz, out_xyz);

  const float radii[3]   = {0.1f, 0.2f, 0.4f};
  const int   ks[3]      = {16, 32, 128};
  const int   mlps[3][3] = {{32, 32, 64}, {64, 64, 128}, {64, 96, 128}};

  int chBase = 0;
  for (int br = 0; br < 3; br++) {
    int  K = ks[br];
    long R = (long)BB * SS * K;
    float r2 = radii[br] * radii[br];

    ballquery_kernel<<<(BB * SS) / 8, 256, 0, stream>>>(xyz, nxyz, gidx, r2, K);
    gather_kernel<<<8192, 256, 0, stream>>>(points, xyz, nxyz, gidx, K, XhA);

    const _Float16* Xp = XhA; int kpad = KPAD1; int cin = 67;
    for (int l = 0; l < 3; l++) {
      int cout = mlps[br][l];
      const float* Wp = (const float*)d_in[2 + (br * 3 + l) * 4 + 0];
      const float* bp = (const float*)d_in[2 + (br * 3 + l) * 4 + 1];
      const float* gp = (const float*)d_in[2 + (br * 3 + l) * 4 + 2];
      const float* be = (const float*)d_in[2 + (br * 3 + l) * 4 + 3];

      prep_w<<<48, 256, 0, stream>>>(Wp, cin, kpad, cout, Wh);

      unsigned gx = (unsigned)(R / 64);
      switch (cout / 32) {
        case 1: gemm_wmma<1><<<gx, 128, 0, stream>>>(Xp, kpad, Wh, bp, Yf); break;
        case 2: gemm_wmma<2><<<gx, 128, 0, stream>>>(Xp, kpad, Wh, bp, Yf); break;
        case 3: gemm_wmma<3><<<gx, 128, 0, stream>>>(Xp, kpad, Wh, bp, Yf); break;
        default: gemm_wmma<4><<<gx, 128, 0, stream>>>(Xp, kpad, Wh, bp, Yf); break;
      }

      stats_partial<<<1024, 128, 0, stream>>>(Yf, R, cout, part);
      float* scl = stats + l * 256;
      float* shf = scl + 128;
      stats_finalize<<<1, 128, 0, stream>>>(part, 1024, gp, be, R, cout, scl, shf);

      if (l < 2) {
        _Float16* Xnext = (l == 0) ? XhB : XhA;   // c0 <= 64, c1 <= 96
        norm_relu_f16<<<8192, 256, 0, stream>>>(Yf, R * cout, scl, shf, cout, Xnext);
        Xp = Xnext; kpad = cout; cin = cout;
      }
    }

    maxpool_kernel<<<BB * SS, 128, 0, stream>>>(Yf, stats + 2 * 256, stats + 2 * 256 + 128,
                                                K, mlps[br][2], out_pts, chBase);
    chBase += mlps[br][2];
  }
  (void)in_sizes; (void)n_in; (void)out_size; (void)ws_size;
}
